// CRTransNet_64072322122342
// MI455X (gfx1250) — matrix operand order repaired and checked by reading, not compiled, not run on hardware
//
#include <hip/hip_runtime.h>
#include <hip/hip_bf16.h>

// ---------------- problem constants ----------------
constexpr int kB   = 2;
constexpr int kS   = 4096;
constexpr int kT   = 2048;
constexpr int kNH  = 10;     // neighbors == local seq len
constexpr int kH   = 10;     // heads
constexpr int kMDL = 160;    // local model dim  (head dim 16)
constexpr int kHDL = 16;
constexpr int kFFL = 32;
constexpr int kMDG = 320;    // grid model dim   (head dim 32)
constexpr int kHD  = 32;
constexpr int kFFG = 1280;
constexpr int kNB  = 64;     // RPE bins
constexpr float kLN100 = 4.6051702f;

// ---------------- workspace layout (bytes) ----------------
constexpr size_t OFF_IDX   = 0;                              // int  [T*NH]
constexpr size_t OFF_XINT  = 81920;                          // f32  [B*T]
constexpr size_t OFF_XSEQ  = 98304;                          // f32  [B*T*NH]
constexpr size_t OFF_QN    = 262144;                         // bf16 [B*H*T*32]
constexpr size_t OFF_KN    = OFF_QN + 2621440;               // bf16
constexpr size_t OFF_VT    = OFF_KN + 2621440;               // bf16 (transposed [B][H][32][T])
constexpr size_t OFF_O     = OFF_VT + 2621440;               // f32  [B*T*320]
// total ~12.8 MB

// ---------------- types ----------------
typedef __attribute__((ext_vector_type(16))) __bf16 v16bf;
typedef __attribute__((ext_vector_type(8)))  __bf16 v8bf;
typedef __attribute__((ext_vector_type(8)))  float  v8f;
typedef __attribute__((ext_vector_type(4)))  unsigned uint4v;
typedef __attribute__((ext_vector_type(8)))  unsigned uint8v;

struct LocalP { const float *Wq,*bq,*Wk,*bk,*Wv,*bv,*Wo,*bo,*ls,*W1,*b1,*W2,*b2; };
struct GridP  { const float *Wq,*bq,*Wk,*bk,*Wv,*bv,*Wo,*bo,*ls,*W1,*b1,*W2,*b2,*g,*be; };

__device__ __forceinline__ __bf16 f2bf(float f) {
  union { float f; unsigned u; } in; in.f = f;
  unsigned r = in.u + 0x7FFFu + ((in.u >> 16) & 1u);   // round-nearest-even
  unsigned short h = (unsigned short)(r >> 16);
  return __builtin_bit_cast(__bf16, h);
}

__device__ __forceinline__ int rpe_bin(float d) {
  // clip(int((d/MAXD + 1) * 0.5 * NBINS), 0, 63); int cast truncates toward 0
  float u = (d * (1.0f / 6.0f) + 1.0f) * 32.0f;
  int i = (int)u;
  return min(max(i, 0), kNB - 1);
}

// Tensor Data Mover: DMA one contiguous 2KB tile (1024 bf16) from global to LDS.
// D# per CDNA5 ISA:
//   group0: dw0 = count(1); dw1 = lds_addr; dw2 = global_addr[31:0];
//           dw3 = global_addr[56:32] | type=2<<30
//   group1: dw0 = data_size=1 (2B) at [17:16], workgroup_mask=0
//           tensor_dim0=1024 @bits79:48, tensor_dim1=1 @bits111:80,
//           tile_dim0=1024 @bits127:112, tile_dim1=1 @bits143:128,
//           tensor_dim0_stride=1024 @bits207:160
// 2-group form (VADDR2/3 = NULL) is valid for <=2D tensors.
__device__ __forceinline__ void tdm_load_2kb(const void* gsrc, unsigned lds_addr) {
  unsigned long long ga = (unsigned long long)(uintptr_t)gsrc;
  uint4v g0 = { 1u, lds_addr, (unsigned)ga,
                (unsigned)((ga >> 32) & 0x01FFFFFFu) | (2u << 30) };
  uint8v g1 = { (1u << 16), (1024u << 16), (1u << 16), (1024u << 16),
                1u, 1024u, 0u, 0u };
  asm volatile("tensor_load_to_lds %0, %1" :: "s"(g0), "s"(g1) : "memory");
}

// ================================================================
// 1) top-NH neighbor search + inverse-distance interpolation
//    one wave per target
// ================================================================
__global__ __launch_bounds__(32)
void k_topk(const float* __restrict__ x, const float* __restrict__ slon,
            const float* __restrict__ slat, const float* __restrict__ tlon,
            const float* __restrict__ tlat, int* __restrict__ idx_out,
            float* __restrict__ xinter, float* __restrict__ xseq) {
  const int t = blockIdx.x;
  const int lane = threadIdx.x;
  const float tl = tlon[t], tp = tlat[t];

  float bd[kNH]; int bi[kNH];
#pragma unroll
  for (int j = 0; j < kNH; ++j) { bd[j] = 3.4e38f; bi[j] = 0x7fffffff; }

  float wsum = 0.f, wx0 = 0.f, wx1 = 0.f;
  for (int s = lane; s < kS; s += 32) {
    float dx = tl - slon[s], dy = tp - slat[s];
    float d2 = dx * dx + dy * dy;
    if (d2 < bd[kNH - 1] || (d2 == bd[kNH - 1] && s < bi[kNH - 1])) {
      bd[kNH - 1] = d2; bi[kNH - 1] = s;
#pragma unroll
      for (int j = kNH - 1; j > 0; --j) {
        if (bd[j] < bd[j - 1] || (bd[j] == bd[j - 1] && bi[j] < bi[j - 1])) {
          float td = bd[j]; bd[j] = bd[j - 1]; bd[j - 1] = td;
          int   ti = bi[j]; bi[j] = bi[j - 1]; bi[j - 1] = ti;
        }
      }
    }
    float w = 1.0f / (sqrtf(d2) + 1e-8f);
    wsum += w; wx0 += w * x[s]; wx1 += w * x[kS + s];
  }
#pragma unroll
  for (int off = 16; off; off >>= 1) {
    wsum += __shfl_xor(wsum, off);
    wx0  += __shfl_xor(wx0, off);
    wx1  += __shfl_xor(wx1, off);
  }

  __shared__ int win[kNH];
  int p = 0;
  for (int r = 0; r < kNH; ++r) {
    float cd = (p < kNH) ? bd[p] : 3.4e38f;
    int   ci = (p < kNH) ? bi[p] : 0x7fffffff;
    int   cl = lane;
#pragma unroll
    for (int off = 16; off; off >>= 1) {
      float od = __shfl_xor(cd, off);
      int   oi = __shfl_xor(ci, off);
      int   ol = __shfl_xor(cl, off);
      if (od < cd || (od == cd && oi < ci)) { cd = od; ci = oi; cl = ol; }
    }
    if (cl == lane) p++;
    if (lane == 0) { idx_out[t * kNH + r] = ci; win[r] = ci; }
  }
  __syncthreads();
  if (lane < kNH) {
    int ii = win[lane];
    xseq[(size_t)(0 * kT + t) * kNH + lane] = x[ii];
    xseq[(size_t)(1 * kT + t) * kNH + lane] = x[kS + ii];
  }
  if (lane == 0) { xinter[t] = wx0 / wsum; xinter[kT + t] = wx1 / wsum; }
}

// ================================================================
// 2) local micro-transformer (2 blocks) entirely in LDS
//    one workgroup per (b,t) sequence of NH=10 tokens
// ================================================================
__global__ __launch_bounds__(128)
void k_local(const float* __restrict__ slon, const float* __restrict__ slat,
             const float* __restrict__ eLonG, const float* __restrict__ eLatG,
             const int* __restrict__ idx, float* __restrict__ xseq,
             LocalP p0, LocalP p1) {
  __shared__ float eLon[kNB * kH], eLat[kNB * kH];
  __shared__ float xt[kNH], nlon[kNH], nlat[kNH];
  __shared__ float q[kNH * kMDL], kk[kNH * kMDL], v[kNH * kMDL], o[kNH * kMDL];
  __shared__ float qs[kNH * kH], ks[kNH * kH];
  __shared__ float att[kH * kNH * kNH];

  const int tid = threadIdx.x;
  const int seq = blockIdx.x;
  const int t = seq % kT;

  for (int i = tid; i < kNB * kH; i += 128) { eLon[i] = eLonG[i]; eLat[i] = eLatG[i]; }
  if (tid < kNH) {
    int ii = idx[t * kNH + tid];
    nlon[tid] = slon[ii]; nlat[tid] = slat[ii];
    xt[tid] = xseq[(size_t)seq * kNH + tid];
  }
  __syncthreads();

  for (int blk = 0; blk < 2; ++blk) {
    LocalP P = blk ? p1 : p0;
    // q,k,v = x * W + b  (din = 1)
    for (int mI = tid; mI < kNH * kMDL; mI += 128) {
      int tok = mI / kMDL, i = mI % kMDL;
      float xv = xt[tok];
      q[mI]  = xv * P.Wq[i] + P.bq[i];
      kk[mI] = xv * P.Wk[i] + P.bk[i];
      v[mI]  = xv * P.Wv[i] + P.bv[i];
    }
    __syncthreads();
    // per (token, head) inverse norms; fold head scale into qs
    for (int u = tid; u < kNH * kH; u += 128) {
      int tok = u / kH, h = u % kH;
      float sq = 0.f, sk = 0.f;
#pragma unroll
      for (int d = 0; d < kHDL; ++d) {
        float a = q[tok * kMDL + h * kHDL + d];  sq += a * a;
        float c = kk[tok * kMDL + h * kHDL + d]; sk += c * c;
      }
      float scale = __expf(fminf(P.ls[h], kLN100));
      qs[u] = rsqrtf(sq + 1e-6f) * scale;
      ks[u] = rsqrtf(sk + 1e-6f);
    }
    __syncthreads();
    // scores + RPE bias
    for (int e = tid; e < kH * kNH * kNH; e += 128) {
      int h = e / (kNH * kNH), rem = e % (kNH * kNH);
      int t1 = rem / kNH, t2 = rem % kNH;
      float dot = 0.f;
#pragma unroll
      for (int d = 0; d < kHDL; ++d)
        dot += q[t1 * kMDL + h * kHDL + d] * kk[t2 * kMDL + h * kHDL + d];
      float sc = dot * qs[t1 * kH + h] * ks[t2 * kH + h];
      float dlon = nlon[t1] - nlon[t2], dlat = nlat[t1] - nlat[t2];
      sc += eLon[rpe_bin(dlon) * kH + h] + eLat[rpe_bin(dlat) * kH + h];
      att[e] = sc;
    }
    __syncthreads();
    // softmax per (head, query)
    for (int r = tid; r < kH * kNH; r += 128) {
      int base = r * kNH;
      float mx = -3.4e38f;
#pragma unroll
      for (int j = 0; j < kNH; ++j) mx = fmaxf(mx, att[base + j]);
      float sm = 0.f;
#pragma unroll
      for (int j = 0; j < kNH; ++j) { float pv = __expf(att[base + j] - mx); att[base + j] = pv; sm += pv; }
      float inv = 1.0f / sm;
#pragma unroll
      for (int j = 0; j < kNH; ++j) att[base + j] *= inv;
    }
    __syncthreads();
    // O = A @ V
    for (int mI = tid; mI < kNH * kMDL; mI += 128) {
      int tok = mI / kMDL, c = mI % kMDL, h = c / kHDL;
      float acc = 0.f;
#pragma unroll
      for (int t2 = 0; t2 < kNH; ++t2)
        acc += att[(h * kNH + tok) * kNH + t2] * v[t2 * kMDL + c];
      o[mI] = acc;
    }
    __syncthreads();
    // Wo (160->1) + residual + FFN (1->32->1, relu,relu) + residual
    if (tid < kNH) {
      float xo = P.bo[0];
      for (int i = 0; i < kMDL; ++i) xo += o[tid * kMDL + i] * P.Wo[i];
      float xr = xt[tid] + xo;
      float fa = P.b2[0];
#pragma unroll
      for (int j = 0; j < kFFL; ++j) {
        float hj = fmaxf(xr * P.W1[j] + P.b1[j], 0.f);
        fa += hj * P.W2[j];
      }
      xt[tid] = xr + fmaxf(fa, 0.f);
    }
    __syncthreads();
  }
  if (tid < kNH) xseq[(size_t)seq * kNH + tid] = xt[tid];
}

// ================================================================
// 3a) grid QKV projection -> normalized bf16 Q/K (scale folded into Q),
//     V stored transposed [B][H][32][T]. one thread per (b,s,h)
// ================================================================
__global__ __launch_bounds__(256)
void k_gproj(const float* __restrict__ xseq, GridP P,
             __bf16* __restrict__ Qn, __bf16* __restrict__ Kn,
             __bf16* __restrict__ Vt) {
  int gid = blockIdx.x * blockDim.x + threadIdx.x;
  if (gid >= kB * kT * kH) return;
  int h = gid % kH;
  int s = (gid / kH) % kT;
  int b = gid / (kH * kT);

  const float* xr = xseq + (size_t)(b * kT + s) * kNH;
  float xv[kNH];
#pragma unroll
  for (int i = 0; i < kNH; ++i) xv[i] = xr[i];

  float scale = __expf(fminf(P.ls[h], kLN100));
  size_t rowQ = ((size_t)(b * kH + h) * kT + s) * kHD;

  float acc[kHD]; float ss;
  // Q
  ss = 0.f;
#pragma unroll
  for (int d = 0; d < kHD; ++d) {
    int col = h * kHD + d;
    float a = P.bq[col];
#pragma unroll
    for (int kkk = 0; kkk < kNH; ++kkk) a += xv[kkk] * P.Wq[kkk * kMDG + col];
    acc[d] = a; ss += a * a;
  }
  {
    float rq = rsqrtf(ss + 1e-6f) * scale;
#pragma unroll
    for (int d = 0; d < kHD; ++d) Qn[rowQ + d] = f2bf(acc[d] * rq);
  }
  // K
  ss = 0.f;
#pragma unroll
  for (int d = 0; d < kHD; ++d) {
    int col = h * kHD + d;
    float a = P.bk[col];
#pragma unroll
    for (int kkk = 0; kkk < kNH; ++kkk) a += xv[kkk] * P.Wk[kkk * kMDG + col];
    acc[d] = a; ss += a * a;
  }
  {
    float rk = rsqrtf(ss + 1e-6f);
#pragma unroll
    for (int d = 0; d < kHD; ++d) Kn[rowQ + d] = f2bf(acc[d] * rk);
  }
  // V (transposed store)
#pragma unroll
  for (int d = 0; d < kHD; ++d) {
    int col = h * kHD + d;
    float a = P.bv[col];
#pragma unroll
    for (int kkk = 0; kkk < kNH; ++kkk) a += xv[kkk] * P.Wv[kkk * kMDG + col];
    Vt[((size_t)(b * kH + h) * kHD + d) * kT + s] = f2bf(a);
  }
}

// ================================================================
// 3b) WMMA flash attention. K tiles staged to LDS by the Tensor Data
//     Mover (double-buffered, TENSORcnt-synchronized); V via vmem with
//     prefetch; RPE bias recomputed in-register from LDS tables.
//     one wave per (b, h, 16-query tile); 32 keys per iteration.
// ================================================================
__global__ __launch_bounds__(32)
void k_attn(const __bf16* __restrict__ Qn, const __bf16* __restrict__ Kn,
            const __bf16* __restrict__ Vt, const float* __restrict__ tlon,
            const float* __restrict__ tlat, const float* __restrict__ eLonG,
            const float* __restrict__ eLatG, float* __restrict__ Obuf) {
  constexpr int QT = kT / 16;  // 128 query tiles
  constexpr int NKT = kT / 32; // 64 key slabs
  const int qt = blockIdx.x % QT;
  const int h  = (blockIdx.x / QT) % kH;
  const int b  = blockIdx.x / (QT * kH);
  const int lane = threadIdx.x;
  const int lg = lane >> 4;      // lane group (0/1)
  const int ln = lane & 15;

  __shared__ float eLonT[kH * kNB], eLatT[kH * kNB];   // [h][bin]
  __shared__ float qlon[16], qlat[16];
  __shared__ __attribute__((aligned(64))) __bf16 Pb[16 * 32];
  __shared__ __attribute__((aligned(64))) __bf16 Kst[2][32 * 32];  // TDM staging

  for (int i = lane; i < kNB * kH; i += 32) {
    int bin = i / kH, hh = i % kH;
    eLonT[hh * kNB + bin] = eLonG[i];
    eLatT[hh * kNB + bin] = eLatG[i];
  }
  if (lane < 16) { qlon[lane] = tlon[qt * 16 + lane]; qlat[lane] = tlat[qt * 16 + lane]; }
  __syncthreads();

  const float* eLonH = &eLonT[h * kNB];
  const float* eLatH = &eLatT[h * kNB];

  const __bf16* Qb = Qn + (size_t)(b * kH + h) * kT * kHD;
  const __bf16* Kb = Kn + (size_t)(b * kH + h) * kT * kHD;
  const __bf16* Vb = Vt + (size_t)(b * kH + h) * kHD * kT;

  // A fragment of Q (16x32 bf16): lanes 0-15 hold K 0-15, lanes 16-31 hold 16-31
  const int qrow = qt * 16 + ln;
  const int k0 = lg ? 16 : 0;
  v8bf qlo = *(const v8bf*)(Qb + (size_t)qrow * kHD + k0);
  v8bf qhi = *(const v8bf*)(Qb + (size_t)qrow * kHD + k0 + 8);
  v16bf aq;
#pragma unroll
  for (int i = 0; i < 8; ++i) { aq[i] = qlo[i]; aq[8 + i] = qhi[i]; }

  // per-row bias constants: bin(d) = clamp((int)(q*32/6 - k*32/6 + 32), 0, 63)
  constexpr float cbin = 32.0f / 6.0f;
  float qsl[8], qsp[8];
#pragma unroll
  for (int e = 0; e < 8; ++e) {
    int r = e + lg * 8;
    qsl[e] = qlon[r] * cbin + 32.0f;
    qsp[e] = qlat[r] * cbin + 32.0f;
  }

  v8f o0 = {}; v8f o1 = {};
  float m[8], l[8];
#pragma unroll
  for (int e = 0; e < 8; ++e) { m[e] = -3.4e38f; l[e] = 0.f; }

  const unsigned kst0 = (unsigned)(uintptr_t)&Kst[0][0];
  const unsigned kst1 = (unsigned)(uintptr_t)&Kst[1][0];
  tdm_load_2kb(Kb, kst0);                  // stage slab 0

  const int ko = lg ? 16 : 0;
  for (int kt = 0; kt < NKT; ++kt) {
    // DMA next slab into the other buffer (wraps harmlessly on last iter)
    const int nxt = (kt + 1 < NKT) ? (kt + 1) : 0;
    tdm_load_2kb(Kb + (size_t)nxt * 1024, (kt & 1) ? kst0 : kst1);
    __builtin_amdgcn_s_wait_tensorcnt(1);  // slab kt resident (TDM in-order)
    asm volatile("" ::: "memory");

    const int kb0 = kt * 32;
    const int ks0 = kb0 + ln, ks1 = kb0 + 16 + ln;
    const __bf16* Ks = (kt & 1) ? &Kst[1][0] : &Kst[0][0];
    // B fragments of K^T (32x16): lane = key column, contiguous 16 hd values
    v16bf bk0 = *(const v16bf*)(Ks + (size_t)ln * kHD + ko);
    v16bf bk1 = *(const v16bf*)(Ks + (size_t)(16 + ln) * kHD + ko);
    __builtin_prefetch(Vb + (size_t)ln * kT + kb0 + 32, 0, 1);  // next V slab

    v8f z = {};
    v8f sc0 = __builtin_amdgcn_wmma_f32_16x16x32_bf16(false, aq, false, bk0,
                                                      (short)0, z, false, false);
    v8f sc1 = __builtin_amdgcn_wmma_f32_16x16x32_bf16(false, aq, false, bk1,
                                                      (short)0, z, false, false);
    float kl0 = tlon[ks0] * cbin, kp0 = tlat[ks0] * cbin;
    float kl1 = tlon[ks1] * cbin, kp1 = tlat[ks1] * cbin;

#pragma unroll
    for (int e = 0; e < 8; ++e) {
      const int r = e + lg * 8;                     // C-layout row this element owns
      int bl0 = min(max((int)(qsl[e] - kl0), 0), kNB - 1);
      int bp0 = min(max((int)(qsp[e] - kp0), 0), kNB - 1);
      int bl1 = min(max((int)(qsl[e] - kl1), 0), kNB - 1);
      int bp1 = min(max((int)(qsp[e] - kp1), 0), kNB - 1);
      sc0[e] += eLonH[bl0] + eLatH[bp0];
      sc1[e] += eLonH[bl1] + eLatH[bp1];
      // online softmax over this 32-key slab (row spread across 16 lanes of group)
      float mm = fmaxf(sc0[e], sc1[e]);
      mm = fmaxf(mm, __shfl_xor(mm, 1));
      mm = fmaxf(mm, __shfl_xor(mm, 2));
      mm = fmaxf(mm, __shfl_xor(mm, 4));
      mm = fmaxf(mm, __shfl_xor(mm, 8));
      float mn = fmaxf(m[e], mm);
      float corr = __expf(m[e] - mn);
      float p0v = __expf(sc0[e] - mn);
      float p1v = __expf(sc1[e] - mn);
      float rs = p0v + p1v;
      rs += __shfl_xor(rs, 1);
      rs += __shfl_xor(rs, 2);
      rs += __shfl_xor(rs, 4);
      rs += __shfl_xor(rs, 8);
      l[e] = l[e] * corr + rs;
      m[e] = mn;
      o0[e] *= corr; o1[e] *= corr;
      Pb[r * 32 + ln]      = f2bf(p0v);
      Pb[r * 32 + 16 + ln] = f2bf(p1v);
    }
    __syncthreads();
    // P as A fragment (16x32) from LDS
    v8bf plo = *(const v8bf*)(&Pb[ln * 32 + k0]);
    v8bf phi = *(const v8bf*)(&Pb[ln * 32 + k0 + 8]);
    v16bf ap;
#pragma unroll
    for (int i = 0; i < 8; ++i) { ap[i] = plo[i]; ap[8 + i] = phi[i]; }
    // V as B fragments: lane = output head-dim column, contiguous 16 keys
    v16bf bv0 = *(const v16bf*)(Vb + (size_t)ln * kT + kb0 + ko);
    v16bf bv1 = *(const v16bf*)(Vb + (size_t)(16 + ln) * kT + kb0 + ko);
    o0 = __builtin_amdgcn_wmma_f32_16x16x32_bf16(false, ap, false, bv0,
                                                 (short)0, o0, false, false);
    o1 = __builtin_amdgcn_wmma_f32_16x16x32_bf16(false, ap, false, bv1,
                                                 (short)0, o1, false, false);
    __syncthreads();
  }

#pragma unroll
  for (int e = 0; e < 8; ++e) {
    const int r = qt * 16 + e + lg * 8;
    float inv = 1.0f / l[e];
    size_t base = (size_t)(b * kT + r) * kMDG + h * kHD;
    Obuf[base + ln]      = o0[e] * inv;
    Obuf[base + 16 + ln] = o1[e] * inv;
  }
}

// ================================================================
// 3c) grid epilogue: O @ Wo + residual, LayerNorm, FFN 10->1280->10
//     one thread per (b,s)
// ================================================================
__global__ __launch_bounds__(256)
void k_gepi(const float* __restrict__ Obuf, GridP P, float* __restrict__ xseq) {
  int gid = blockIdx.x * blockDim.x + threadIdx.x;
  if (gid >= kB * kT) return;
  const float* orow = Obuf + (size_t)gid * kMDG;
  float* xp = xseq + (size_t)gid * kNH;

  float acc[kNH];
#pragma unroll
  for (int j = 0; j < kNH; ++j) acc[j] = P.bo[j];
  for (int i = 0; i < kMDG; ++i) {
    float ov = orow[i];
    const float* wr = P.Wo + i * kNH;
#pragma unroll
    for (int j = 0; j < kNH; ++j) acc[j] += ov * wr[j];
  }
  float xr[kNH];
  float mn = 0.f;
#pragma unroll
  for (int j = 0; j < kNH; ++j) { xr[j] = xp[j] + acc[j]; mn += xr[j]; }
  mn *= (1.0f / kNH);
  float var = 0.f;
#pragma unroll
  for (int j = 0; j < kNH; ++j) { float d = xr[j] - mn; var += d * d; }
  var *= (1.0f / kNH);
  float rstd = rsqrtf(var + 1e-5f);
#pragma unroll
  for (int j = 0; j < kNH; ++j) xr[j] = (xr[j] - mn) * rstd * P.g[j] + P.be[j];

  float f2[kNH];
#pragma unroll
  for (int j = 0; j < kNH; ++j) f2[j] = P.b2[j];
  for (int jj = 0; jj < kFFG; ++jj) {
    float hh = P.b1[jj];
#pragma unroll
    for (int kkk = 0; kkk < kNH; ++kkk) hh += xr[kkk] * P.W1[kkk * kFFG + jj];
    hh = fmaxf(hh, 0.f);
    const float* w2 = P.W2 + jj * kNH;
#pragma unroll
    for (int kkk = 0; kkk < kNH; ++kkk) f2[kkk] += hh * w2[kkk];
  }
#pragma unroll
  for (int j = 0; j < kNH; ++j) xp[j] = xr[j] + fmaxf(f2[j], 0.f);
}

// ================================================================
// 4) output head: x_inter + relu(xg @ Wout + bout)
// ================================================================
__global__ __launch_bounds__(256)
void k_final(const float* __restrict__ xseq, const float* __restrict__ xinter,
             const float* __restrict__ Wout, const float* __restrict__ bout,
             float* __restrict__ out) {
  int gid = blockIdx.x * blockDim.x + threadIdx.x;
  if (gid >= kB * kT) return;
  float s = bout[0];
#pragma unroll
  for (int j = 0; j < kNH; ++j) s += xseq[(size_t)gid * kNH + j] * Wout[j];
  out[gid] = xinter[gid] + fmaxf(s, 0.f);
}

// ================================================================
extern "C" void kernel_launch(void* const* d_in, const int* in_sizes, int n_in,
                              void* d_out, int out_size, void* d_ws, size_t ws_size,
                              hipStream_t stream) {
  (void)in_sizes; (void)n_in; (void)out_size; (void)ws_size;
  auto F = [&](int i) { return (const float*)d_in[i]; };

  const float* x    = F(0);
  const float* slon = F(1);
  const float* slat = F(2);
  const float* tlon = F(3);
  const float* tlat = F(4);
  const float* eLon = F(5);
  const float* eLat = F(6);

  auto LP = [&](int blk) {
    int b0 = 7 + 13 * blk;
    LocalP p{F(b0), F(b0 + 1), F(b0 + 2), F(b0 + 3), F(b0 + 4), F(b0 + 5),
             F(b0 + 6), F(b0 + 7), F(b0 + 8), F(b0 + 9), F(b0 + 10),
             F(b0 + 11), F(b0 + 12)};
    return p;
  };
  auto GP = [&](int blk) {
    int b0 = 33 + 15 * blk;
    GridP p{F(b0), F(b0 + 1), F(b0 + 2), F(b0 + 3), F(b0 + 4), F(b0 + 5),
            F(b0 + 6), F(b0 + 7), F(b0 + 8), F(b0 + 9), F(b0 + 10),
            F(b0 + 11), F(b0 + 12), F(b0 + 13), F(b0 + 14)};
    return p;
  };
  const float* Wout = F(78);
  const float* bout = F(79);

  char* ws = (char*)d_ws;
  int*    idxw   = (int*)(ws + OFF_IDX);
  float*  xinter = (float*)(ws + OFF_XINT);
  float*  xseq   = (float*)(ws + OFF_XSEQ);
  __bf16* Qn     = (__bf16*)(ws + OFF_QN);
  __bf16* Kn     = (__bf16*)(ws + OFF_KN);
  __bf16* Vt     = (__bf16*)(ws + OFF_VT);
  float*  Obuf   = (float*)(ws + OFF_O);

  k_topk<<<kT, 32, 0, stream>>>(x, slon, slat, tlon, tlat, idxw, xinter, xseq);
  k_local<<<kB * kT, 128, 0, stream>>>(slon, slat, eLon, eLat, idxw, xseq, LP(0), LP(1));

  for (int g = 0; g < 3; ++g) {
    GridP P = GP(g);
    k_gproj<<<(kB * kT * kH + 255) / 256, 256, 0, stream>>>(xseq, P, Qn, Kn, Vt);
    k_attn<<<kB * kH * (kT / 16), 32, 0, stream>>>(Qn, Kn, Vt, tlon, tlat, eLon, eLat, Obuf);
    k_gepi<<<(kB * kT + 255) / 256, 256, 0, stream>>>(Obuf, P, xseq);
  }
  k_final<<<(kB * kT + 255) / 256, 256, 0, stream>>>(xseq, xinter, Wout, bout, (float*)d_out);
}
